// PKM_12412455485500
// MI455X (gfx1250) — matrix operand order repaired and checked by reading, not compiled, not run on hardware
//
#include <hip/hip_runtime.h>
#include <hip/hip_bf16.h>

typedef __attribute__((ext_vector_type(16))) _Float16 v16h;
typedef __attribute__((ext_vector_type(8)))  float    v8f;
typedef unsigned int v4u __attribute__((ext_vector_type(4)));
typedef int          v8i __attribute__((ext_vector_type(8)));
typedef int          v4i __attribute__((ext_vector_type(4)));

#define TOPK   32
#define DIM    1024
#define DHALF  512
#define NKEYS  1024
#define NTOK   16384      // 128 * 128
#define NTILES 1024       // NTOK / 16
#define LN_EPS 1e-5f

// ---------------------------------------------------------------------------
// Kernel 1: LayerNorm + f16 convert + pack into WMMA A-fragment order.
// A-frag layout (f16 16x32, wave32): lane<16 -> row=lane, k in {0..7,16..23};
// lane>=16 -> row=lane-16, k in {8..15,24..31}; element e -> kk = (e<8?e:e+8).
// qn layout: [tile][plane][kchunk][lane*16 + e]
// ---------------------------------------------------------------------------
__global__ __launch_bounds__(512)
void pack_qn_kernel(const float* __restrict__ x,
                    const float* __restrict__ gamma,
                    const float* __restrict__ beta,
                    _Float16* __restrict__ qn)
{
    __shared__ float s_mu[16][2];
    __shared__ float s_rs[16][2];

    const int T    = blockIdx.x;        // token tile
    const int tid  = threadIdx.x;
    const int w    = tid >> 5;          // wave 0..15
    const int lane = tid & 31;

    // Phase 1: wave w computes LN stats for token T*16+w (both halves)
    {
        const int t = T * 16 + w;
        const float* row = x + (size_t)t * DIM;
        #pragma unroll
        for (int p = 0; p < 2; ++p) {
            const float4* src = (const float4*)(row + p * DHALF + lane * 16);
            float s = 0.f, ss = 0.f;
            #pragma unroll
            for (int i = 0; i < 4; ++i) {
                float4 v = src[i];
                s  += v.x + v.y + v.z + v.w;
                ss += v.x * v.x + v.y * v.y + v.z * v.z + v.w * v.w;
            }
            #pragma unroll
            for (int off = 16; off > 0; off >>= 1) {
                s  += __shfl_xor(s, off);
                ss += __shfl_xor(ss, off);
            }
            float mu  = s * (1.f / DHALF);
            float var = ss * (1.f / DHALF) - mu * mu;
            if (lane == 0) { s_mu[w][p] = mu; s_rs[w][p] = rsqrtf(var + LN_EPS); }
        }
    }
    __syncthreads();

    // Phase 2: each wave writes 2 of the 32 fragments (2 planes x 16 chunks)
    const int m  = lane & 15;
    const int hi = lane >> 4;
    #pragma unroll
    for (int ff = 0; ff < 2; ++ff) {
        const int f = w + ff * 16;      // 0..31
        const int p = f >> 4;
        const int c = f & 15;
        const int t = T * 16 + m;
        const float mu = s_mu[m][p];
        const float rs = s_rs[m][p];
        const float* src = x + (size_t)t * DIM + p * DHALF + c * 32;
        v16h frag;
        #pragma unroll
        for (int e = 0; e < 16; ++e) {
            int kk = (e < 8) ? e : e + 8;
            int kl = kk + 8 * hi;       // 0..31 within chunk
            int d  = c * 32 + kl;       // 0..511 within half
            float q = (src[kl] - mu) * rs * gamma[d] + beta[d];
            frag[e] = (_Float16)q;
        }
        *(v16h*)(qn + (((size_t)T * 2 + p) * 16 + c) * 512 + lane * 16) = frag;
    }
}

// ---------------------------------------------------------------------------
// Kernel 2: pack keys into WMMA B-fragment order.
// B-frag (f16 32x16): lane n = lane&15 (column), k = 8*(lane>>4) + (e<8?e:e+8).
// kf layout: [plane][ntile][kchunk][lane*16 + e]
// ---------------------------------------------------------------------------
__global__ __launch_bounds__(256)
void pack_keys_kernel(const float* __restrict__ keys,
                      _Float16* __restrict__ kf)
{
    const int gtid = blockIdx.x * 256 + threadIdx.x;
    const int w    = gtid >> 5;         // fragment id 0..2047
    const int lane = gtid & 31;

    const int p   = w >> 10;            // plane
    const int rem = w & 1023;
    const int Tn  = rem >> 4;           // key tile 0..63
    const int c   = rem & 15;           // k chunk
    const int n   = Tn * 16 + (lane & 15);
    const int hi  = lane >> 4;

    const float* src = keys + ((size_t)n * 2 + p) * DHALF + c * 32;
    v16h frag;
    #pragma unroll
    for (int e = 0; e < 16; ++e) frag[e] = (_Float16)src[((e < 8) ? e : e + 8) + 8 * hi];
    *(v16h*)(kf + (((size_t)p * 64 + Tn) * 16 + c) * 512 + lane * 16) = frag;
}

// ---------------------------------------------------------------------------
// Kernel 3: fused WMMA GEMM + per-token top-32 + 32x32 combine top-32.
// Block = 1024 threads (32 waves) handles a 32-token tile (2 WMMA M-tiles).
// A fragments are DMA'd into LDS by the Tensor Data Mover (one 2D tile:
// 2 rows x 16 KB, 32 KB row stride), issued by wave 0 and tracked with
// TENSORcnt.  Each wave owns one 16-key N-tile per half and feeds each
// loaded B fragment to TWO WMMAs (both M-tiles).
// LDS: A frags 32 KB + fp32 scores 64 KB + running tops 16 KB = 112 KB.
// ---------------------------------------------------------------------------
__global__ __launch_bounds__(1024)
void gemm_topk_kernel(const _Float16* __restrict__ qn,
                      const _Float16* __restrict__ kf,
                      float* __restrict__ outS,
                      int*   __restrict__ outI)
{
    __shared__ __attribute__((aligned(128))) _Float16 lds_A[2 * 16 * 512]; // 32 KB
    __shared__ float lds_scores[32 * 512];                                 // 64 KB
    __shared__ float topS[2][32][32];                                      // 8 KB
    __shared__ int   topI[2][32][32];                                      // 8 KB

    const int blk  = blockIdx.x;        // 32-token block
    const int tid  = threadIdx.x;
    const int w    = tid >> 5;          // wave 0..31
    const int lane = tid & 31;

    for (int p = 0; p < 2; ++p) {
        // ---- TDM: stage A fragments (both M-tiles) for plane p into LDS ----
        // Source rows: qn element offset ((blk*2+mt)*2+p)*8192, mt = 0,1
        //   -> base + mt*16384 elements: 2D tile, dim0 = 8192 elems (16 KB),
        //      dim1 = 2, dim0_stride = 16384 elems, data_size = 2 bytes.
        if (w == 0) {
            unsigned long long ga =
                (unsigned long long)(uintptr_t)(qn + (((size_t)blk * 4) + p) * 8192);
            unsigned int lo = (unsigned int)(uintptr_t)(&lds_A[0]); // LDS offset
            v4u g0 = { 1u,                                   // count=1, user mode
                       lo,                                   // lds_addr (bytes)
                       (unsigned int)ga,                     // global_addr[31:0]
                       (unsigned int)((ga >> 32) & 0x01FFFFFFu) | (2u << 30) }; // +type=2
            v8i g1 = { (int)(1u << 16),        // data_size = 1 (2 bytes)
                       (int)(8192u << 16),     // tensor_dim0[15:0] in [31:16]
                       (int)(2u << 16),        // tensor_dim0 hi=0 | tensor_dim1=2
                       (int)(8192u << 16),     // tensor_dim1 hi=0 | tile_dim0=8192
                       2,                      // tile_dim1=2, tile_dim2=0
                       16384,                  // tensor_dim0_stride lo32 (elems)
                       0, 0 };                 // stride hi / dim1_stride
            v4i z4 = { 0, 0, 0, 0 };
#if __clang_major__ >= 23
            v8i z8 = { 0, 0, 0, 0, 0, 0, 0, 0 };
            __builtin_amdgcn_tensor_load_to_lds(g0, g1, z4, z4, z8, 0);
#else
            __builtin_amdgcn_tensor_load_to_lds(g0, g1, z4, z4, 0);
#endif
            __builtin_amdgcn_s_wait_tensorcnt(0);
        }
        __syncthreads();

        for (int h = 0; h < 2; ++h) {
            // ---- GEMM: 32 N-tiles this half; wave w owns N-tile w ----
            {
                const int ntile = h * 32 + w;   // 0..63
                const _Float16* bbase =
                    kf + (((size_t)p * 64 + ntile) * 16) * 512;
                v8f acc0 = {0.f, 0.f, 0.f, 0.f, 0.f, 0.f, 0.f, 0.f};
                v8f acc1 = {0.f, 0.f, 0.f, 0.f, 0.f, 0.f, 0.f, 0.f};
                #pragma unroll
                for (int c = 0; c < 16; ++c) {
                    v16h b  = *(const v16h*)(bbase + (size_t)c * 512 + lane * 16);
                    v16h a0 = *(const v16h*)(lds_A + c * 512 + lane * 16);
                    v16h a1 = *(const v16h*)(lds_A + 8192 + c * 512 + lane * 16);
                    acc0 = __builtin_amdgcn_wmma_f32_16x16x32_f16(
                        false, a0, false, b, (short)0, acc0, false, false);
                    acc1 = __builtin_amdgcn_wmma_f32_16x16x32_f16(
                        false, a1, false, b, (short)0, acc1, false, false);
                }
                // C layout: lane<16 -> N=lane, M=r ; lane>=16 -> N=lane-16, M=r+8
                const int ncol  = w * 16 + (lane & 15);    // column within half
                const int mbase = (lane >> 4) * 8;
                #pragma unroll
                for (int r = 0; r < 8; ++r) {
                    lds_scores[(mbase + r) * 512 + ncol]      = acc0[r];
                    lds_scores[(16 + mbase + r) * 512 + ncol] = acc1[r];
                }
            }
            __syncthreads();

            // ---- top-32 over this half (+ merge running top-32) ----
            {
                const int m = w;        // token row (0..31) handled by this wave
                float cand[17]; int cidx[17];
                #pragma unroll
                for (int j = 0; j < 16; ++j) {
                    cand[j] = lds_scores[m * 512 + lane * 16 + j];
                    cidx[j] = h * 512 + lane * 16 + j;     // global key idx
                }
                int ncand = 16;
                cand[16] = -__builtin_inff(); cidx[16] = -1;
                if (h == 1) {
                    cand[16] = topS[p][m][lane];
                    cidx[16] = topI[p][m][lane];
                    ncand = 17;
                }
                float resS = 0.f; int resI = 0;
                for (int k = 0; k < TOPK; ++k) {
                    float best = -__builtin_inff(); int bj = 0;
                    #pragma unroll
                    for (int j = 0; j < 17; ++j)
                        if (j < ncand && cand[j] > best) { best = cand[j]; bj = j; }
                    float v = best; int gi = cidx[bj]; int src = lane;
                    #pragma unroll
                    for (int off = 16; off > 0; off >>= 1) {
                        float ov  = __shfl_xor(v, off);
                        int   ogi = __shfl_xor(gi, off);
                        int   osr = __shfl_xor(src, off);
                        if (ov > v || (ov == v && ogi < gi)) { v = ov; gi = ogi; src = osr; }
                    }
                    if (lane == src) {      // invalidate winner (indices unique)
                        #pragma unroll
                        for (int j = 0; j < 17; ++j)
                            if (cidx[j] == gi) cand[j] = -__builtin_inff();
                    }
                    if (lane == k) { resS = v; resI = gi; }
                }
                topS[p][m][lane] = resS;
                topI[p][m][lane] = resI;
            }
            __syncthreads();
        }
        __syncthreads();   // safe to overwrite lds_A for next plane
    }

    // ---- combine: top-32 of sx[i]+sy[j] over 32x32, wave w -> token w ----
    {
        const int m = w;
        const float sx = topS[0][m][lane];
        const int   ix = topI[0][m][lane];
        float cand[32]; int cidx[32];
        #pragma unroll
        for (int j = 0; j < 32; ++j) {
            cand[j] = sx + topS[1][m][j];
            cidx[j] = ix * NKEYS + topI[1][m][j];
        }
        float resS = 0.f; int resI = 0;
        for (int k = 0; k < TOPK; ++k) {
            float best = -__builtin_inff(); int bj = 0;
            #pragma unroll
            for (int j = 0; j < 32; ++j)
                if (cand[j] > best) { best = cand[j]; bj = j; }
            float v = best; int gi = cidx[bj]; int src = lane;
            #pragma unroll
            for (int off = 16; off > 0; off >>= 1) {
                float ov  = __shfl_xor(v, off);
                int   ogi = __shfl_xor(gi, off);
                int   osr = __shfl_xor(src, off);
                if (ov > v || (ov == v && ogi < gi)) { v = ov; gi = ogi; src = osr; }
            }
            if (lane == src) {
                #pragma unroll
                for (int j = 0; j < 32; ++j)
                    if (cidx[j] == gi) cand[j] = -__builtin_inff();
            }
            if (lane == k) { resS = v; resI = gi; }
        }
        const int t = blk * 32 + m;
        outS[(size_t)t * TOPK + lane] = resS;
        outI[(size_t)t * TOPK + lane] = resI;
    }
}

// ---------------------------------------------------------------------------
extern "C" void kernel_launch(void* const* d_in, const int* in_sizes, int n_in,
                              void* d_out, int out_size, void* d_ws, size_t ws_size,
                              hipStream_t stream) {
    (void)in_sizes; (void)n_in; (void)out_size; (void)ws_size;

    const float* x     = (const float*)d_in[0];   // (16384, 1024)
    const float* keys  = (const float*)d_in[1];   // (1024, 2, 512)
    const float* gamma = (const float*)d_in[2];   // (512,)
    const float* beta  = (const float*)d_in[3];   // (512,)

    // Workspace: qn f16 fragments (32 MB) + key f16 fragments (2 MB)
    _Float16* qn = (_Float16*)d_ws;
    _Float16* kf = qn + (size_t)NTOK * DIM;

    float* outS = (float*)d_out;                          // 16384*32 f32
    int*   outI = (int*)d_out + (size_t)NTOK * TOPK;      // 16384*32 i32

    pack_qn_kernel  <<<NTILES,     512,  0, stream>>>(x, gamma, beta, qn);
    pack_keys_kernel<<<256,        256,  0, stream>>>(keys, kf);
    gemm_topk_kernel<<<NTILES / 2, 1024, 0, stream>>>(qn, kf, outS, outI);
}
